// CT_metrics_system_70042326663202
// MI455X (gfx1250) — matrix operand order repaired and checked by reading, not compiled, not run on hardware
//
#include <hip/hip_runtime.h>
#include <hip/hip_bf16.h>

// ---------------------------------------------------------------------------
// Problem constants (fixed by the reference): N=512 memories, M=512 queries,
// D=1024 features, H=512 hidden1, H2=256 hidden2.
// ---------------------------------------------------------------------------
constexpr int N_  = 512;
constexpr int M_  = 512;
constexpr int D_  = 1024;
constexpr int H_  = 512;
constexpr int H2_ = 256;
constexpr float NEG_SLOPE = 0.01f;

constexpr int TM = 32;    // (n,m) pairs per block (2 row-tiles of 16)
constexpr int KC = 128;   // K-chunk staged in LDS per barrier pair (4 WMMA steps)

typedef __attribute__((ext_vector_type(16))) __bf16        bf16x16;
typedef __attribute__((ext_vector_type(4)))  __bf16        bf16x4;
typedef __attribute__((ext_vector_type(8)))  float         f32x8;
typedef __attribute__((ext_vector_type(4)))  unsigned int  u32x4;

union FragU { bf16x16 bf; u32x4 u[2]; };

// A-matrix fragment (16x32 bf16, ISA 7.12.2 layout):
// lane L (0..15) = row L, K chunks [0..7] and [16..23];
// lane L+16      = row L, K chunks [8..15] and [24..31].
__device__ __forceinline__ bf16x16 load_fragA(const __bf16* base, int strideElems, int lane) {
    const int row  = lane & 15;
    const int koff = (lane >> 4) * 8;
    const __bf16* p = base + row * strideElems + koff;
    FragU f;
    f.u[0] = *reinterpret_cast<const u32x4*>(p);        // K = koff .. koff+7
    f.u[1] = *reinterpret_cast<const u32x4*>(p + 16);   // K = koff+16 .. koff+23
    return f.bf;
}

// B-matrix fragment (32x16 bf16): lane L (0..15) = column L, K [0..7] and
// [16..23]; lane L+16 = column L, K [8..15] and [24..31]. With W stored row-major
// [cols][K], a column of B = a row of W -> one contiguous 64-byte run.
__device__ __forceinline__ bf16x16 load_fragB(const __bf16* base, int strideElems, int lane) {
    const int col  = lane & 15;
    const int koff = (lane >> 4) * 8;
    const __bf16* p = base + col * strideElems + koff;
    FragU f;
    f.u[0] = *reinterpret_cast<const u32x4*>(p);
    f.u[1] = *reinterpret_cast<const u32x4*>(p + 16);
    return f.bf;
}

__device__ __forceinline__ float leaky(float v) {
    return v > 0.f ? v : NEG_SLOPE * v;
}

// ---------------------------------------------------------------------------
// K0: fp32 -> bf16 weight conversion (runs once per launch; trivial traffic)
// ---------------------------------------------------------------------------
__global__ void cvt_bf16_kernel(const float* __restrict__ in, __bf16* __restrict__ out, int n) {
    int i = blockIdx.x * blockDim.x + threadIdx.x;
    if (i < n) out[i] = (__bf16)in[i];
}

// ---------------------------------------------------------------------------
// K1: fully fused  diff2 -> GEMM1 -> leaky -> GEMM2 -> leaky -> dot(w3)
// One block = 32 (n,m) pairs (fixed n, 32 consecutive m). 256 threads, 8 waves.
// Phase A: per wave 2 row-tiles x 4 h-col-tiles (8 accumulators); one barrier
// pair per 128-wide K chunk (4 WMMA K-steps, B fragments reused across rows).
// ---------------------------------------------------------------------------
__global__ __launch_bounds__(256)
void fused_scorer_kernel(const float* __restrict__ Ma, const float* __restrict__ Q,
                         const __bf16* __restrict__ W1b, const float* __restrict__ b1,
                         const __bf16* __restrict__ W2b, const float* __restrict__ b2,
                         const float* __restrict__ W3,  const float* __restrict__ b3,
                         float* __restrict__ d_mat, float* __restrict__ cost_mat)
{
    __shared__ __bf16 diff_lds[TM * KC];     // 8 KB: staged A chunk for GEMM1
    __shared__ __bf16 h1_lds[TM * H_];       // 32 KB: fused h1 tile (A for GEMM2)
    __shared__ float  d_acc[TM];
    __shared__ float  cost_acc[TM];

    const int tid  = threadIdx.x;
    const int lane = tid & 31;
    const int wave = tid >> 5;

    const int tilesPerRow = M_ / TM;                 // 16
    const int n  = blockIdx.x / tilesPerRow;
    const int m0 = (blockIdx.x % tilesPerRow) * TM;

    if (tid < TM) { d_acc[tid] = 0.f; cost_acc[tid] = 0.f; }

    // diff2 generation: thread t owns row (t>>3), 16 consecutive k-cols.
    const int grow = tid >> 3;                       // pair row 0..31
    const int gcol = (tid & 7) * 16;                 // k-col base within chunk
    const float* mrow = Ma + (size_t)n * D_;
    const float* qrow = Q  + (size_t)(m0 + grow) * D_;
    float costp = 0.f;

    // ------------------ Phase A: h1 = leaky(diff2 @ W1^T + b1) ------------------
    f32x8 acc[2][4];
    const f32x8 zf = {0.f,0.f,0.f,0.f,0.f,0.f,0.f,0.f};
#pragma unroll
    for (int r = 0; r < 2; ++r)
#pragma unroll
        for (int t = 0; t < 4; ++t) acc[r][t] = zf;

    for (int k0 = 0; k0 < D_; k0 += KC) {
        __syncthreads();                              // LDS chunk free to rewrite
#pragma unroll
        for (int cc = 0; cc < 16; cc += 4) {
            const float4 mv = *(const float4*)(mrow + k0 + gcol + cc);
            const float4 qv = *(const float4*)(qrow + k0 + gcol + cc);
            float e0 = mv.x - qv.x, e1 = mv.y - qv.y, e2 = mv.z - qv.z, e3 = mv.w - qv.w;
            e0 *= e0; e1 *= e1; e2 *= e2; e3 *= e3;
            costp += (e0 + e1) + (e2 + e3);           // cost = sum_d diff2, for free
            bf16x4 bv;
            bv[0] = (__bf16)e0; bv[1] = (__bf16)e1; bv[2] = (__bf16)e2; bv[3] = (__bf16)e3;
            *reinterpret_cast<bf16x4*>(&diff_lds[grow * KC + gcol + cc]) = bv;
        }
        __syncthreads();                              // chunk visible to all waves

#pragma unroll
        for (int kk = 0; kk < KC; kk += 32) {
            const bf16x16 a0 = load_fragA(diff_lds + kk,           KC, lane); // rows 0..15
            const bf16x16 a1 = load_fragA(diff_lds + 16 * KC + kk, KC, lane); // rows 16..31
#pragma unroll
            for (int t = 0; t < 4; ++t) {
                const int h0 = wave * 64 + t * 16;
                const bf16x16 b = load_fragB(W1b + (size_t)h0 * D_ + k0 + kk, D_, lane);
                acc[0][t] = __builtin_amdgcn_wmma_f32_16x16x32_bf16(
                                false, a0, false, b, (short)0, acc[0][t], false, false);
                acc[1][t] = __builtin_amdgcn_wmma_f32_16x16x32_bf16(
                                false, a1, false, b, (short)0, acc[1][t], false, false);
            }
        }
    }

    // bias + LeakyReLU, spill h1 tile to LDS as bf16 for GEMM2.
    // C layout: lane%16 -> column h, VGPR j -> pair row j + 8*(lane/16).
    {
        const int psub = (lane >> 4) * 8;
#pragma unroll
        for (int r = 0; r < 2; ++r) {
#pragma unroll
            for (int t = 0; t < 4; ++t) {
                const int h    = wave * 64 + t * 16 + (lane & 15);
                const float bh = b1[h];
#pragma unroll
                for (int j = 0; j < 8; ++j) {
                    const float v = leaky(acc[r][t][j] + bh);
                    h1_lds[(r * 16 + psub + j) * H_ + h] = (__bf16)v;
                }
            }
        }
    }
    atomicAdd(&cost_acc[grow], costp);                // ds_add_f32
    __syncthreads();                                  // h1 tile complete

    // ------------------ Phase B: h2 = leaky(h1 @ W2^T + b2) ------------------
    f32x8 acc2[2][2];
    acc2[0][0] = zf; acc2[0][1] = zf; acc2[1][0] = zf; acc2[1][1] = zf;
    for (int k0 = 0; k0 < H_; k0 += 32) {
        const bf16x16 a0 = load_fragA(h1_lds + k0,            H_, lane);
        const bf16x16 a1 = load_fragA(h1_lds + 16 * H_ + k0,  H_, lane);
#pragma unroll
        for (int t = 0; t < 2; ++t) {
            const int c0 = wave * 32 + t * 16;
            const bf16x16 b = load_fragB(W2b + (size_t)c0 * H_ + k0, H_, lane);
            acc2[0][t] = __builtin_amdgcn_wmma_f32_16x16x32_bf16(
                             false, a0, false, b, (short)0, acc2[0][t], false, false);
            acc2[1][t] = __builtin_amdgcn_wmma_f32_16x16x32_bf16(
                             false, a1, false, b, (short)0, acc2[1][t], false, false);
        }
    }

    // ------------------ Phase C: d = -(h2 . w3 + b3), straight from registers ----
    float pd[2][8];
#pragma unroll
    for (int r = 0; r < 2; ++r)
#pragma unroll
        for (int j = 0; j < 8; ++j) pd[r][j] = 0.f;
#pragma unroll
    for (int t = 0; t < 2; ++t) {
        const int col  = wave * 32 + t * 16 + (lane & 15);
        const float bc = b2[col];
        const float w3 = W3[col];
#pragma unroll
        for (int r = 0; r < 2; ++r)
#pragma unroll
            for (int j = 0; j < 8; ++j)
                pd[r][j] += leaky(acc2[r][t][j] + bc) * w3;
    }
    {
        const int psub = (lane >> 4) * 8;
#pragma unroll
        for (int r = 0; r < 2; ++r)
#pragma unroll
            for (int j = 0; j < 8; ++j)
                atomicAdd(&d_acc[r * 16 + psub + j], pd[r][j]);
    }
    __syncthreads();

    if (tid < TM) {
        const size_t o = (size_t)n * M_ + m0 + tid;
        d_mat[o]    = -(d_acc[tid] + b3[0]);
        cost_mat[o] = cost_acc[tid];
    }
}

// ---------------------------------------------------------------------------
// K2: per-column softmax over n + weighted sums.  One block per m.
// ---------------------------------------------------------------------------
__global__ __launch_bounds__(256)
void softmax_score_kernel(const float* __restrict__ d_mat, const float* __restrict__ cost_mat,
                          const int* __restrict__ nfg_p, float* __restrict__ out)
{
    __shared__ float red[256];
    const int m = blockIdx.x;
    const int t = threadIdx.x;
    const int nfg = *nfg_p;

    const float v0 = d_mat[(size_t)t * M_ + m];
    const float v1 = d_mat[(size_t)(t + 256) * M_ + m];

    // max over n
    red[t] = fmaxf(v0, v1);
    __syncthreads();
    for (int s = 128; s > 0; s >>= 1) {
        if (t < s) red[t] = fmaxf(red[t], red[t + s]);
        __syncthreads();
    }
    const float mx = red[0];
    __syncthreads();

    const float e0 = __expf(v0 - mx);
    const float e1 = __expf(v1 - mx);

    // sum of exp
    red[t] = e0 + e1;
    __syncthreads();
    for (int s = 128; s > 0; s >>= 1) {
        if (t < s) red[t] += red[t + s];
        __syncthreads();
    }
    const float denom = red[0];
    __syncthreads();

    const float w0 = cost_mat[(size_t)t * M_ + m] * e0;
    const float w1 = cost_mat[(size_t)(t + 256) * M_ + m] * e1;

    // full weighted sum
    red[t] = w0 + w1;
    __syncthreads();
    for (int s = 128; s > 0; s >>= 1) {
        if (t < s) red[t] += red[t + s];
        __syncthreads();
    }
    const float tot = red[0];
    __syncthreads();

    // foreground (n < N_fg) weighted sum
    red[t] = (t < nfg ? w0 : 0.f) + (t + 256 < nfg ? w1 : 0.f);
    __syncthreads();
    for (int s = 128; s > 0; s >>= 1) {
        if (t < s) red[t] += red[t + s];
        __syncthreads();
    }
    if (t == 0) {
        out[m]      = tot / denom;       // score
        out[M_ + m] = red[0] / denom;    // score_fg
    }
}

// ---------------------------------------------------------------------------
// Launch: K0 (weight cvt) -> K1 (fused WMMA scorer) -> K2 (softmax reduce)
// Workspace layout: W1 bf16 (1 MB) | W2 bf16 (256 KB) | d (1 MB) | cost (1 MB)
// ---------------------------------------------------------------------------
extern "C" void kernel_launch(void* const* d_in, const int* in_sizes, int n_in,
                              void* d_out, int out_size, void* d_ws, size_t ws_size,
                              hipStream_t stream) {
    const float* Ma = (const float*)d_in[0];
    const float* Q  = (const float*)d_in[1];
    const float* W1 = (const float*)d_in[2];
    const float* b1 = (const float*)d_in[3];
    const float* W2 = (const float*)d_in[4];
    const float* b2 = (const float*)d_in[5];
    const float* W3 = (const float*)d_in[6];
    const float* b3 = (const float*)d_in[7];
    const int*  nfg = (const int*)d_in[8];

    char* ws = (char*)d_ws;
    __bf16* W1b      = (__bf16*)ws;                                    // H*D
    __bf16* W2b      = (__bf16*)(ws + (size_t)H_ * D_ * 2);            // H2*H
    float*  d_mat    = (float*)(ws + (size_t)H_ * D_ * 2 + (size_t)H2_ * H_ * 2);
    float*  cost_mat = d_mat + (size_t)N_ * M_;

    const int nW1 = H_ * D_;
    const int nW2 = H2_ * H_;
    cvt_bf16_kernel<<<(nW1 + 255) / 256, 256, 0, stream>>>(W1, W1b, nW1);
    cvt_bf16_kernel<<<(nW2 + 255) / 256, 256, 0, stream>>>(W2, W2b, nW2);

    const int nBlocks = (N_ * M_) / TM;   // 8192 pair-tiles
    fused_scorer_kernel<<<nBlocks, 256, 0, stream>>>(
        Ma, Q, W1b, b1, W2b, b2, W3, b3, d_mat, cost_mat);

    softmax_score_kernel<<<M_, 256, 0, stream>>>(d_mat, cost_mat, nfg, (float*)d_out);
}
